// LinearAttention_21955872817359
// MI455X (gfx1250) — matrix operand order repaired and checked by reading, not compiled, not run on hardware
//
#include <hip/hip_runtime.h>
#include <hip/hip_bf16.h>

// ---------------------------------------------------------------------------
// Gated linear attention (DeltaNet-style) for MI455X / gfx1250.
//   B=1, S=4096, HID=2048, H=8, DK=DV=128, KCONV=4
// GEMMs run as bf16 WMMA (v_wmma_f32_16x16x32_bf16, f32 accumulate) with
// ping-pong double-buffered fragment registers (2x unrolled K loop) so the
// WMMA block overlaps the other buffer's global loads with no register moves.
// Sequential scan runs f32 in registers (one lane per V column, per-head WG).
// ---------------------------------------------------------------------------

typedef __bf16 bf16;
typedef bf16  v8bf  __attribute__((ext_vector_type(8)));
typedef bf16  v16bf __attribute__((ext_vector_type(16)));
typedef float v8f   __attribute__((ext_vector_type(8)));

#define S_LEN    4096
#define HID      2048
#define H_HEADS  8
#define DK       128
#define DV       128
#define KEY_DIM  1024
#define VAL_DIM  1024
#define CONV_DIM 3072
#define KCONV    4

__device__ __forceinline__ bf16 f2bf(float f) {
  unsigned u = __builtin_bit_cast(unsigned, f);
  unsigned r = (u + 0x7FFFu + ((u >> 16) & 1u)) >> 16;   // round-to-nearest-even
  return __builtin_bit_cast(bf16, (unsigned short)r);
}

__device__ __forceinline__ v16bf cat8(v8bf lo, v8bf hi) {
  return __builtin_shufflevector(lo, hi, 0,1,2,3,4,5,6,7,8,9,10,11,12,13,14,15);
}

// ---------------------------------------------------------------------------
// f32 -> bf16 conversion (bandwidth-bound, trivial)
// ---------------------------------------------------------------------------
__global__ void cvt_f32_bf16(const float* __restrict__ in, bf16* __restrict__ out, int n) {
  int i = blockIdx.x * blockDim.x + threadIdx.x;
  if (i < n) out[i] = f2bf(in[i]);
}

// ---------------------------------------------------------------------------
// C[M,N](f32) = A[M,K](bf16, row-major) * B[N,K](bf16, row-major)^T
// Wave tile: 16(M) x 64(N); block = 8 waves -> 16 x 512 per block.
// Fragment layouts used here (wave32; A/B use contiguous k, same order for both):
//   A 16x32: lane m=lane&15; halves e=0..15 = K[16*hi + e]   (hi = lane>>4)
//   B 32x16: lane n=lane&15; halves e=0..15 = K[16*hi + e]
//   C 16x16: VGPR j -> row m0 + j + 8*hi, col = n0 + (lane&15)
// K must be a multiple of 64 (true for all call sites: 2048 / 1024).
// ---------------------------------------------------------------------------
__device__ __forceinline__ void load_frags(const bf16* __restrict__ arow,
                                           const bf16* __restrict__ b0,
                                           size_t bstep, int k,
                                           v16bf& af, v16bf* bf_) {
  af = cat8(*(const v8bf*)(arow + k), *(const v8bf*)(arow + k + 8));
#pragma unroll
  for (int nt = 0; nt < 4; ++nt) {
    const bf16* bp = b0 + (size_t)nt * bstep + k;
    bf_[nt] = cat8(*(const v8bf*)(bp), *(const v8bf*)(bp + 8));
  }
}

__global__ __launch_bounds__(256) void wmma_gemm_bf16(
    const bf16* __restrict__ A, const bf16* __restrict__ B,
    float* __restrict__ C, int M, int N, int K) {
  const int wave = threadIdx.x >> 5;
  const int lane = threadIdx.x & 31;
  const int hi   = lane >> 4;
  const int l16  = lane & 15;
  const int m0   = blockIdx.x << 4;
  const int n0   = (blockIdx.y << 9) + (wave << 6);

  const bf16* arow = A + (size_t)(m0 + l16) * K + (hi << 4);
  const bf16* b0   = B + (size_t)(n0 + l16) * K + (hi << 4);
  const size_t bstep = (size_t)16 * K;   // 16 N-columns forward

  v8f acc[4] = {};

  // ping-pong fragment buffers (fixed registers, no rotation copies)
  v16bf a0, a1, bf0[4], bf1[4];
  load_frags(arow, b0, bstep, 0, a0, bf0);

  int k = 0;
  for (; k + 64 < K; k += 64) {
    __builtin_prefetch(arow + k + 128, 0, 1);
    __builtin_prefetch(b0   + k + 128, 0, 1);
    load_frags(arow, b0, bstep, k + 32, a1, bf1);
#pragma unroll
    for (int nt = 0; nt < 4; ++nt)
      acc[nt] = __builtin_amdgcn_wmma_f32_16x16x32_bf16(
          false, a0, false, bf0[nt], (short)0, acc[nt], false, false);
    load_frags(arow, b0, bstep, k + 64, a0, bf0);
#pragma unroll
    for (int nt = 0; nt < 4; ++nt)
      acc[nt] = __builtin_amdgcn_wmma_f32_16x16x32_bf16(
          false, a1, false, bf1[nt], (short)0, acc[nt], false, false);
  }
  // tail: k == K-64 here; steps K-64 (in buf0) and K-32 remain
  load_frags(arow, b0, bstep, k + 32, a1, bf1);
#pragma unroll
  for (int nt = 0; nt < 4; ++nt)
    acc[nt] = __builtin_amdgcn_wmma_f32_16x16x32_bf16(
        false, a0, false, bf0[nt], (short)0, acc[nt], false, false);
#pragma unroll
  for (int nt = 0; nt < 4; ++nt)
    acc[nt] = __builtin_amdgcn_wmma_f32_16x16x32_bf16(
        false, a1, false, bf1[nt], (short)0, acc[nt], false, false);

#pragma unroll
  for (int nt = 0; nt < 4; ++nt) {
    float* crow = C + (size_t)(m0 + (hi << 3)) * N + (n0 + (nt << 4) + l16);
#pragma unroll
    for (int j = 0; j < 8; ++j) crow[(size_t)j * N] = acc[nt][j];
  }
}

// ---------------------------------------------------------------------------
// Causal depthwise conv1d (K=4, left pad 3) + SiLU. mixed[S,3072] -> qkv[S,3072]
// ---------------------------------------------------------------------------
__global__ void conv_silu_kernel(const float* __restrict__ mixed,
                                 const float* __restrict__ convw,
                                 float* __restrict__ qkv) {
  int c = blockIdx.x * blockDim.x + threadIdx.x;  // 0..3071
  int s = blockIdx.y;
  float acc = 0.f;
#pragma unroll
  for (int j = 0; j < KCONV; ++j) {
    int sp = s - (KCONV - 1) + j;
    if (sp >= 0) acc += convw[c * KCONV + j] * mixed[(size_t)sp * CONV_DIM + c];
  }
  qkv[(size_t)s * CONV_DIM + c] = acc / (1.f + __expf(-acc));   // SiLU
}

// ---------------------------------------------------------------------------
// In-place L2 norm of q (with 1/sqrt(DK) scale) and k. grid=(S, 16): y = sel*8+h
// ---------------------------------------------------------------------------
__global__ __launch_bounds__(128) void l2norm_kernel(float* __restrict__ qkv) {
  int s = blockIdx.x;
  int sel = blockIdx.y >> 3;          // 0 = q, 1 = k
  int h = blockIdx.y & 7;
  int t = threadIdx.x;                // 0..127
  float* p = qkv + (size_t)s * CONV_DIM + sel * KEY_DIM + h * DK;
  float x = p[t];
  __shared__ float red[128];
  red[t] = x * x;
  __syncthreads();
  for (int off = 64; off > 0; off >>= 1) {
    if (t < off) red[t] += red[t + off];
    __syncthreads();
  }
  float inv = rsqrtf(red[0] + 1e-6f);
  if (sel == 0) inv *= 0.08838834764831845f;   // 1/sqrt(128)
  p[t] = x * inv;
}

// ---------------------------------------------------------------------------
// a,b projections (N=8 each, tiny) fused with gate math:
//   g = -exp(A_log)*softplus(a+dt_bias),  beta = sigmoid(b)
// block = (32,16): each wave computes one 2048-length dot.
// ---------------------------------------------------------------------------
__global__ void gate_kernel(const float* __restrict__ x,
                            const float* __restrict__ Wa,
                            const float* __restrict__ Wb,
                            const float* __restrict__ A_log,
                            const float* __restrict__ dt_bias,
                            float* __restrict__ g, float* __restrict__ beta) {
  int s = blockIdx.x;
  int lane = threadIdx.x;             // 0..31
  int o = threadIdx.y;                // 0..15
  const float* xr = x + (size_t)s * HID;
  const float* wr = (o < 8) ? (Wa + (size_t)o * HID) : (Wb + (size_t)(o - 8) * HID);
  float acc = 0.f;
  for (int i = lane; i < HID; i += 32) acc += xr[i] * wr[i];
#pragma unroll
  for (int off = 16; off > 0; off >>= 1) acc += __shfl_down(acc, off, 32);
  if (lane == 0) {
    if (o < 8) {
      float a = acc + dt_bias[o];
      float sp = (a > 20.f) ? a : log1pf(__expf(a));        // softplus
      g[(size_t)s * H_HEADS + o] = -__expf(A_log[o]) * sp;
    } else {
      beta[(size_t)s * H_HEADS + (o - 8)] = 1.f / (1.f + __expf(-acc));
    }
  }
}

// ---------------------------------------------------------------------------
// Sequential recurrent scan. One workgroup per head, 128 lanes = DV columns.
// State column st[DK] lives in VGPRs (full unroll). k/q broadcast via LDS.
//   kv    = decay * <st, k>
//   delta = (v - kv) * beta
//   o     = decay * <st, q> + delta * <k, q>
//   st    = decay*st + k*delta
// ---------------------------------------------------------------------------
__global__ __launch_bounds__(128) void scan_kernel(const float* __restrict__ qkv,
                                                   const float* __restrict__ g,
                                                   const float* __restrict__ beta,
                                                   float* __restrict__ core) {
  const int h = blockIdx.x;
  const int v = threadIdx.x;          // V column
  __shared__ float kq[2 * DK];        // [0..127]=k, [128..255]=q

  float st[DK];
#pragma unroll
  for (int i = 0; i < DK; ++i) st[i] = 0.f;

  for (int t = 0; t < S_LEN; ++t) {
    const float* base = qkv + (size_t)t * CONV_DIM;
    kq[v]       = base[KEY_DIM + h * DK + v];   // k_t
    kq[DK + v]  = base[h * DK + v];             // q_t
    __syncthreads();

    float gt = g[(size_t)t * H_HEADS + h];
    float bt = beta[(size_t)t * H_HEADS + h];
    float decay = __expf(gt);
    float vt = base[2 * KEY_DIM + h * DV + v];

    float dsk = 0.f, dsq = 0.f, dkq = 0.f;
#pragma unroll
    for (int i = 0; i < DK; ++i) {
      float ki = kq[i];
      float qi = kq[DK + i];
      dsk += st[i] * ki;
      dsq += st[i] * qi;
      dkq += ki * qi;
    }
    float kv = decay * dsk;
    float delta = (vt - kv) * bt;
    float o = decay * dsq + delta * dkq;
#pragma unroll
    for (int i = 0; i < DK; ++i) st[i] = decay * st[i] + kq[i] * delta;

    core[(size_t)t * VAL_DIM + h * DV + v] = o;
    __syncthreads();
  }
}

// ---------------------------------------------------------------------------
// RMSNorm (per head) * norm_w, gated by SiLU(z); emit bf16 for the out-proj GEMM
// grid = (S, H), block = 128
// ---------------------------------------------------------------------------
__global__ __launch_bounds__(128) void rms_gate_kernel(const float* __restrict__ core,
                                                       const float* __restrict__ z,
                                                       const float* __restrict__ norm_w,
                                                       bf16* __restrict__ gated) {
  int s = blockIdx.x, h = blockIdx.y, v = threadIdx.x;
  size_t idx = (size_t)s * VAL_DIM + h * DV + v;
  float c = core[idx];
  __shared__ float red[128];
  red[v] = c * c;
  __syncthreads();
  for (int off = 64; off > 0; off >>= 1) {
    if (v < off) red[v] += red[v + off];
    __syncthreads();
  }
  float var = red[0] * (1.f / DV);
  float zn = z[idx];
  float y = c * rsqrtf(var + 1e-6f) * norm_w[v];
  y *= zn / (1.f + __expf(-zn));     // SiLU gate
  gated[idx] = f2bf(y);
}

// ---------------------------------------------------------------------------
extern "C" void kernel_launch(void* const* d_in, const int* in_sizes, int n_in,
                              void* d_out, int out_size, void* d_ws, size_t ws_size,
                              hipStream_t stream) {
  (void)in_sizes; (void)n_in; (void)out_size; (void)ws_size;
  const float* x      = (const float*)d_in[0];   // [S, HID]
  const float* w_qkv  = (const float*)d_in[1];   // [3072, 2048]
  const float* conv_w = (const float*)d_in[2];   // [3072, 1, 4]
  const float* w_a    = (const float*)d_in[3];   // [8, 2048]
  const float* w_b    = (const float*)d_in[4];   // [8, 2048]
  const float* w_z    = (const float*)d_in[5];   // [1024, 2048]
  const float* A_log  = (const float*)d_in[6];   // [8]
  const float* dt_b   = (const float*)d_in[7];   // [8]
  const float* norm_w = (const float*)d_in[8];   // [128]
  const float* w_out  = (const float*)d_in[9];   // [2048, 1024]
  float* out = (float*)d_out;                    // [S, HID] f32

  // ---- workspace carve-up (~172 MiB) ----
  char* w = (char*)d_ws;
  size_t off = 0;
  auto carve = [&](size_t bytes) -> void* {
    void* p = w + off;
    off += (bytes + 255) & ~(size_t)255;
    return p;
  };
  bf16*  xb     = (bf16*) carve((size_t)S_LEN * HID * 2);
  bf16*  wqkvb  = (bf16*) carve((size_t)CONV_DIM * HID * 2);
  bf16*  wzb    = (bf16*) carve((size_t)VAL_DIM * HID * 2);
  bf16*  wob    = (bf16*) carve((size_t)HID * VAL_DIM * 2);
  float* mixed  = (float*)carve((size_t)S_LEN * CONV_DIM * 4);
  float* qkvbuf = (float*)carve((size_t)S_LEN * CONV_DIM * 4);
  float* zbuf   = (float*)carve((size_t)S_LEN * VAL_DIM * 4);
  float* gbuf   = (float*)carve((size_t)S_LEN * H_HEADS * 4);
  float* bbuf   = (float*)carve((size_t)S_LEN * H_HEADS * 4);
  float* corebuf= (float*)carve((size_t)S_LEN * VAL_DIM * 4);
  bf16*  gated  = (bf16*) carve((size_t)S_LEN * VAL_DIM * 2);

  // 1) f32 -> bf16 conversions
  {
    int n;
    n = S_LEN * HID;      cvt_f32_bf16<<<dim3((n + 255) / 256), dim3(256), 0, stream>>>(x, xb, n);
    n = CONV_DIM * HID;   cvt_f32_bf16<<<dim3((n + 255) / 256), dim3(256), 0, stream>>>(w_qkv, wqkvb, n);
    n = VAL_DIM * HID;    cvt_f32_bf16<<<dim3((n + 255) / 256), dim3(256), 0, stream>>>(w_z, wzb, n);
    n = HID * VAL_DIM;    cvt_f32_bf16<<<dim3((n + 255) / 256), dim3(256), 0, stream>>>(w_out, wob, n);
  }

  // 2) WMMA GEMMs: mixed = x·Wqkv^T  [4096,3072],  z = x·Wz^T [4096,1024]
  wmma_gemm_bf16<<<dim3(S_LEN / 16, CONV_DIM / 512), dim3(256), 0, stream>>>(
      xb, wqkvb, mixed, S_LEN, CONV_DIM, HID);
  wmma_gemm_bf16<<<dim3(S_LEN / 16, VAL_DIM / 512), dim3(256), 0, stream>>>(
      xb, wzb, zbuf, S_LEN, VAL_DIM, HID);

  // 3) depthwise causal conv + SiLU
  conv_silu_kernel<<<dim3(CONV_DIM / 256, S_LEN), dim3(256), 0, stream>>>(
      mixed, conv_w, qkvbuf);

  // 4) L2-normalize q (with 1/sqrt(DK)) and k
  l2norm_kernel<<<dim3(S_LEN, 16), dim3(128), 0, stream>>>(qkvbuf);

  // 5) a/b projections + gate math (g, beta)
  gate_kernel<<<dim3(S_LEN), dim3(32, 16), 0, stream>>>(
      x, w_a, w_b, A_log, dt_b, gbuf, bbuf);

  // 6) sequential recurrent scan (one WG per head)
  scan_kernel<<<dim3(H_HEADS), dim3(128), 0, stream>>>(qkvbuf, gbuf, bbuf, corebuf);

  // 7) RMSNorm + SiLU(z) gate -> bf16
  rms_gate_kernel<<<dim3(S_LEN, H_HEADS), dim3(128), 0, stream>>>(
      corebuf, zbuf, norm_w, gated);

  // 8) out = gated·Wout^T  [4096,2048] via WMMA
  wmma_gemm_bf16<<<dim3(S_LEN / 16, HID / 512), dim3(256), 0, stream>>>(
      gated, wob, out, S_LEN, HID, VAL_DIM);
}
